// GQA_28114855920089
// MI455X (gfx1250) — matrix operand order repaired and checked by reading, not compiled, not run on hardware
//
#include <hip/hip_runtime.h>
#include <hip/hip_bf16.h>

// ---------------------------------------------------------------------------
// CDNA5 (gfx1250) GQA forward: bf16 WMMA (v_wmma_f32_16x16x32_bf16), wave32,
// async global->LDS staging (global_load_async_to_lds_b128 / ASYNCcnt).
// ---------------------------------------------------------------------------

typedef __attribute__((ext_vector_type(16))) __bf16 v16bf;
typedef __attribute__((ext_vector_type(8)))  __bf16 v8bf;
typedef __attribute__((ext_vector_type(8)))  float  v8f;

static __device__ __forceinline__ v8f wmma_bf16(v16bf a, v16bf b, v8f c) {
  // D = A(16x32 bf16) * B(32x16 bf16) + C(16x16 f32)
  return __builtin_amdgcn_wmma_f32_16x16x32_bf16(
      /*neg_a=*/false, a, /*neg_b=*/false, b,
      /*c_mod=*/(short)0, c, /*reuse_a=*/false, /*reuse_b=*/false);
}

// 32-bit LDS byte offset of a __shared__ object (generic -> AS3 -> int).
static __device__ __forceinline__ unsigned lds_off32(const void* p) {
  return (unsigned)(size_t)(const __attribute__((address_space(3))) void*)p;
}

// Async DMA: 16 bytes global -> LDS, per lane. Tracked by ASYNCcnt.
static __device__ __forceinline__ void async_copy_b128(const void* gsrc,
                                                       const void* ldst) {
  asm volatile("global_load_async_to_lds_b128 %0, %1, off"
               :: "v"(lds_off32(ldst)), "v"(gsrc)
               : "memory");
}
static __device__ __forceinline__ void wait_async() {
  asm volatile("s_wait_asynccnt 0x0" ::: "memory");
}

// A-fragment (16x32, MxK): lane<16 holds row M=lane, K in {0..7, 16..23};
// lane>=16 holds row M=lane-16, K in {8..15, 24..31}.  base = tile origin,
// ld = row stride in halves (rows must be 16B aligned => ld % 8 == 0).
static __device__ __forceinline__ v16bf frag_a(const __bf16* base, int ld) {
  const int lane = threadIdx.x & 31;
  const int m = lane & 15, hi = lane >> 4;
  const __bf16* p = base + (size_t)m * ld;
  v8bf lo = *(const v8bf*)(p + hi * 8);
  v8bf hh = *(const v8bf*)(p + 16 + hi * 8);
  return __builtin_shufflevector(lo, hh, 0,1,2,3,4,5,6,7,8,9,10,11,12,13,14,15);
}

// B-fragment (32x16, KxN) read from a K-transposed LDS tile laid out [n][k]:
// lane-half 0 -> K {0..7, 16..23}, lane-half 1 -> K {8..15, 24..31}, N = lane&15.
static __device__ __forceinline__ v16bf frag_b(const __bf16* base, int ld) {
  const int lane = threadIdx.x & 31;
  const int n = lane & 15, hi = lane >> 4;
  const __bf16* p = base + (size_t)n * ld + hi * 8;
  v8bf lo = *(const v8bf*)(p);
  v8bf hh = *(const v8bf*)(p + 16);
  return __builtin_shufflevector(lo, hh, 0,1,2,3,4,5,6,7,8,9,10,11,12,13,14,15);
}

// ---------------------------------------------------------------------------
// GEMM + bias:  C[M,N] = A[M,K] @ W[K,N] + bias[N]
// 128x128 workgroup tile, BK=32, 8 waves (2x4), 64x32 per wave, 8 WMMA/step.
// bf16 A tiles are staged with async DMA; fp32 A and W convert through VGPRs.
// ---------------------------------------------------------------------------
#define LDA 40   // 32 + 8 pad, halves (row = 80B, multiple of 16B)
#define LDB 40

template <typename AT, typename OT>
__global__ __launch_bounds__(256) void gemm_bias_kernel(
    const AT* __restrict__ A, const float* __restrict__ W,
    const float* __restrict__ bias, OT* __restrict__ C,
    int M, int N, int K) {
  (void)M;
  __shared__ __align__(16) __bf16 As[128 * LDA];   // [m][k]
  __shared__ __align__(16) __bf16 Bs[128 * LDB];   // [n][k] (transposed)

  const int tid  = threadIdx.x;
  const int lane = tid & 31;
  const int wid  = tid >> 5;
  const int wm   = wid >> 2;          // 0..1
  const int wn   = wid & 3;           // 0..3
  const int m0   = blockIdx.y * 128;
  const int n0   = blockIdx.x * 128;

  v8f acc[4][2] = {};

  const int ar = tid >> 1, ac = (tid & 1) * 16;      // A: 128 rows x 32 cols
  const int wr = tid >> 3, wc = (tid & 7) * 16;      // W: 32 rows x 128 cols

  for (int k0 = 0; k0 < K; k0 += 32) {
    __syncthreads();
    // --- stage A tile ---
    {
      const AT* src = A + (size_t)(m0 + ar) * K + k0 + ac;
      __bf16* dst = &As[ar * LDA + ac];
      if constexpr (sizeof(AT) == 2) {
        // already bf16: direct async DMA global->LDS (2 x 16B per thread)
        async_copy_b128(src, dst);
        async_copy_b128(src + 8, dst + 8);
      } else {
#pragma unroll
        for (int i = 0; i < 16; ++i) dst[i] = (__bf16)(float)src[i];
        if (k0 + 32 < K) __builtin_prefetch(src + 32, 0, 3);
      }
    }
    // --- stage W tile transposed: Bs[n][k] (convert fp32 -> bf16) ---
    {
      const float* src = W + (size_t)(k0 + wr) * N + n0 + wc;
#pragma unroll
      for (int i = 0; i < 16; ++i) Bs[(wc + i) * LDB + wr] = (__bf16)src[i];
      if (k0 + 32 < K) __builtin_prefetch(src + (size_t)32 * N, 0, 3);
    }
    if constexpr (sizeof(AT) == 2) wait_async();
    __syncthreads();

    v16bf af[4], bfm[2];
#pragma unroll
    for (int mt = 0; mt < 4; ++mt)
      af[mt] = frag_a(&As[(wm * 64 + mt * 16) * LDA], LDA);
#pragma unroll
    for (int nt = 0; nt < 2; ++nt)
      bfm[nt] = frag_b(&Bs[(wn * 32 + nt * 16) * LDB], LDB);
#pragma unroll
    for (int mt = 0; mt < 4; ++mt)
#pragma unroll
      for (int nt = 0; nt < 2; ++nt)
        acc[mt][nt] = wmma_bf16(af[mt], bfm[nt], acc[mt][nt]);
  }

  // --- epilogue: bias + store (C layout: M = r + 8*(lane>=16), N = lane&15)
  const int hi = lane >> 4, nl = lane & 15;
#pragma unroll
  for (int mt = 0; mt < 4; ++mt) {
#pragma unroll
    for (int nt = 0; nt < 2; ++nt) {
      const int col = n0 + wn * 32 + nt * 16 + nl;
      const float bv = bias[col];
#pragma unroll
      for (int r = 0; r < 8; ++r) {
        const int row = m0 + wm * 64 + mt * 16 + hi * 8 + r;
        C[(size_t)row * N + col] = (OT)(acc[mt][nt][r] + bv);
      }
    }
  }
}

// ---------------------------------------------------------------------------
// Flash attention (GQA):  O = softmax(Q K^T / sqrt(64)) V
// One workgroup per (b, h, 128-query block); wave w owns 16 query rows.
// KV streamed in blocks of 32; K tile staged via async DMA; online softmax;
// per-wave P converted C-layout -> A-layout through LDS.
// ---------------------------------------------------------------------------
#define A_S   2048
#define A_HID 2048
#define A_KVD 512
#define A_HD  64
#define LDK   72   // K tile [kv][d]: 64 + 8 pad
#define LDV   40   // V tile [d][kv]: 32 + 8 pad
#define LDP   40   // P tile [m][kv]: 32 + 8 pad

__global__ __launch_bounds__(256) void attn_kernel(
    const __bf16* __restrict__ Q, const __bf16* __restrict__ Kb,
    const __bf16* __restrict__ Vb, __bf16* __restrict__ O) {
  __shared__ __align__(16) __bf16 Ks[32 * LDK];        // [kv][d]
  __shared__ __align__(16) __bf16 Vs[64 * LDV];        // [d][kv] (transposed)
  __shared__ __align__(16) __bf16 Ps[8 * 16 * LDP];    // per-wave [16][kv]

  const int tid  = threadIdx.x;
  const int lane = tid & 31;
  const int wid  = tid >> 5;
  const int hi   = lane >> 4, nl = lane & 15;

  const int h  = blockIdx.y;
  const int b  = blockIdx.z;
  const int q0 = blockIdx.x * 128;
  const int g  = h >> 2;                    // NH/NG = 4 heads per KV group
  const int qrow0 = q0 + wid * 16;

  const __bf16* qptr = Q  + ((size_t)b * A_S) * A_HID + h * A_HD;
  const __bf16* kptr = Kb + ((size_t)b * A_S) * A_KVD + g * A_HD;
  const __bf16* vptr = Vb + ((size_t)b * A_S) * A_KVD + g * A_HD;

  // --- load Q fragments once, straight from global (A layout) ---
  v16bf qf[2];
  {
    const __bf16* p = qptr + (size_t)(qrow0 + (lane & 15)) * A_HID;
#pragma unroll
    for (int kt = 0; kt < 2; ++kt) {
      v8bf lo = *(const v8bf*)(p + kt * 32 + hi * 8);
      v8bf hh = *(const v8bf*)(p + kt * 32 + 16 + hi * 8);
      qf[kt] = __builtin_shufflevector(lo, hh,
                 0,1,2,3,4,5,6,7,8,9,10,11,12,13,14,15);
    }
  }

  float m_run[8], l_run[8];
  v8f oacc[4] = {};
#pragma unroll
  for (int r = 0; r < 8; ++r) { m_run[r] = -1e30f; l_run[r] = 0.f; }

  const int krow = tid >> 3, kcol = (tid & 7) * 8;   // staging: 32 x 64 halves
  const float SCALE = 0.125f;                        // 1/sqrt(64)

  for (int kb = 0; kb < A_S; kb += 32) {
    __syncthreads();
    // --- stage K [kv][d] via async DMA; V transposed [d][kv] via VGPRs ---
    {
      async_copy_b128(kptr + (size_t)(kb + krow) * A_KVD + kcol,
                      &Ks[krow * LDK + kcol]);
      v8bf vd = *(const v8bf*)(vptr + (size_t)(kb + krow) * A_KVD + kcol);
#pragma unroll
      for (int j = 0; j < 8; ++j) Vs[(kcol + j) * LDV + krow] = vd[j];
      if (kb + 32 < A_S)
        __builtin_prefetch(vptr + (size_t)(kb + 32 + krow) * A_KVD + kcol, 0, 3);
    }
    wait_async();
    __syncthreads();

    // --- scores: S = Q @ K^T  (2 n-tiles of 16 kv, 2 k-steps over d) ---
    v8f s[2];
#pragma unroll
    for (int nt = 0; nt < 2; ++nt) {
      v8f a = {};
      a = wmma_bf16(qf[0], frag_b(&Ks[(nt * 16) * LDK + 0],  LDK), a);
      a = wmma_bf16(qf[1], frag_b(&Ks[(nt * 16) * LDK + 32], LDK), a);
      s[nt] = a;
    }

    // --- online softmax (row M = r + 8*hi lives across the 16-lane half) ---
    __bf16* pw = &Ps[wid * 16 * LDP];
#pragma unroll
    for (int r = 0; r < 8; ++r) {
      float v0 = s[0][r] * SCALE, v1 = s[1][r] * SCALE;
      float mx = fmaxf(v0, v1);
#pragma unroll
      for (int off = 8; off; off >>= 1) mx = fmaxf(mx, __shfl_xor(mx, off, 32));
      const float mnew  = fmaxf(m_run[r], mx);
      const float alpha = __expf(m_run[r] - mnew);
      const float p0 = __expf(v0 - mnew), p1 = __expf(v1 - mnew);
      float rs = p0 + p1;
#pragma unroll
      for (int off = 8; off; off >>= 1) rs += __shfl_xor(rs, off, 32);
      l_run[r] = l_run[r] * alpha + rs;
      m_run[r] = mnew;
#pragma unroll
      for (int nt = 0; nt < 4; ++nt) oacc[nt][r] *= alpha;
      // C-layout -> LDS so P can be re-read in A-layout
      pw[(r + 8 * hi) * LDP + nl]      = (__bf16)p0;
      pw[(r + 8 * hi) * LDP + 16 + nl] = (__bf16)p1;
    }

    // --- O += P @ V  (4 d-tiles of 16) ---
    v16bf pf = frag_a(pw, LDP);
#pragma unroll
    for (int nt = 0; nt < 4; ++nt)
      oacc[nt] = wmma_bf16(pf, frag_b(&Vs[(nt * 16) * LDV], LDV), oacc[nt]);
  }

  // --- epilogue: O /= l, store bf16 into [B,S,HID] at head offset ---
  __bf16* optr = O + ((size_t)b * A_S + qrow0) * A_HID + h * A_HD;
#pragma unroll
  for (int nt = 0; nt < 4; ++nt)
#pragma unroll
    for (int r = 0; r < 8; ++r)
      optr[(size_t)(hi * 8 + r) * A_HID + nt * 16 + nl] =
          (__bf16)(oacc[nt][r] / l_run[r]);
}

// ---------------------------------------------------------------------------
// Host-side orchestration
// ---------------------------------------------------------------------------
extern "C" void kernel_launch(void* const* d_in, const int* in_sizes, int n_in,
                              void* d_out, int out_size, void* d_ws, size_t ws_size,
                              hipStream_t stream) {
  (void)in_sizes; (void)n_in; (void)out_size; (void)ws_size;
  constexpr int B = 2, S = 2048, HID = 2048, KVD = 512, NH = 32;
  constexpr int M = B * S;

  const float* hs = (const float*)d_in[0];
  const float* Wq = (const float*)d_in[1]; const float* bq = (const float*)d_in[2];
  const float* Wk = (const float*)d_in[3]; const float* bk = (const float*)d_in[4];
  const float* Wv = (const float*)d_in[5]; const float* bv = (const float*)d_in[6];
  const float* Wo = (const float*)d_in[7]; const float* bo = (const float*)d_in[8];

  __bf16* qb = (__bf16*)d_ws;                       // [M, HID]
  __bf16* kb = qb + (size_t)M * HID;                // [M, KVD]
  __bf16* vb = kb + (size_t)M * KVD;                // [M, KVD]
  __bf16* ab = vb + (size_t)M * KVD;                // [M, HID]

  dim3 blk(256);
  gemm_bias_kernel<float, __bf16><<<dim3(HID / 128, M / 128), blk, 0, stream>>>(
      hs, Wq, bq, qb, M, HID, HID);
  gemm_bias_kernel<float, __bf16><<<dim3(KVD / 128, M / 128), blk, 0, stream>>>(
      hs, Wk, bk, kb, M, KVD, HID);
  gemm_bias_kernel<float, __bf16><<<dim3(KVD / 128, M / 128), blk, 0, stream>>>(
      hs, Wv, bv, vb, M, KVD, HID);
  attn_kernel<<<dim3(S / 128, NH, B), blk, 0, stream>>>(qb, kb, vb, ab);
  gemm_bias_kernel<__bf16, float><<<dim3(HID / 128, M / 128), blk, 0, stream>>>(
      ab, Wo, bo, (float*)d_out, M, HID, HID);
}
